// MAK_27401891348771
// MI455X (gfx1250) — matrix operand order repaired and checked by reading, not compiled, not run on hardware
//
#include <hip/hip_runtime.h>

// ---------------------------------------------------------------------------
// CDNA5 (gfx1250, wave32) implementation.
// Pipeline (all on `stream`, deterministic):
//   K0 prepW   : Wsum[o, i*32+o'] = sum_h W1[o*128+i*4+h, o']  (f32 -> bf16),
//                stored directly in WMMA A-fragment order (per-lane packed)
//   K1 gemm1   : z0 = W0 @ y           (WMMA bf16, f32 acc) + BN1 partials
//   K2 bnstats : reduce partials -> scale/shift for BN1
//   K3 gemm2   : z1 = Wm @ lrelu(bn(z0))  (WMMA) + BN2 partials
//   K4 bnstats : -> BN2 scale/shift
//   K5 main    : out_raw = Wsum @ q, q[i*32+o',p] = x[i,p]*t1[o',p]
//                (q built per-step in registers, Wsum async-staged in LDS)
//   K6 bnstats : -> BN3 scale/shift
//   K7 epilog  : lrelu(bn(out_raw) + x)  -> d_out (B,Cout,N,K)
// ---------------------------------------------------------------------------

typedef __attribute__((ext_vector_type(16))) __bf16 v16bf;
typedef __attribute__((ext_vector_type(8)))  float  v8f;
typedef __attribute__((ext_vector_type(4)))  unsigned int u32x4;
typedef __attribute__((ext_vector_type(4)))  int    i32x4;
typedef __attribute__((ext_vector_type(4)))  float  f32x4;

union BF16x16 {
  v16bf v;
  u32x4 u[2];
  __bf16 e[16];
};

#define DEV __device__ __forceinline__

#if defined(__has_builtin)
#if __has_builtin(__builtin_amdgcn_global_load_async_to_lds_b128) && \
    __has_builtin(__builtin_amdgcn_s_wait_asynccnt)
#define USE_ASYNC_LDS 1
#endif
#endif
#ifndef USE_ASYNC_LDS
#define USE_ASYNC_LDS 0
#endif

constexpr int B_    = 2;
constexpr int CIN   = 32;
constexpr int COUT  = 32;
constexpr int CF    = 64;
constexpr int NDIM  = 1024;
constexpr int KDIM  = 20;
constexpr int NK    = NDIM * KDIM;   // 20480
constexpr int P     = B_ * NK;       // 40960 pixels
constexpr float EPS   = 1e-5f;
constexpr float SLOPE = 0.2f;

constexpr int NWG_GEMM = 160;        // workgroups for GEMM kernels (256 thr = 8 waves)
constexpr int WS_ELEMS = 32 * 1024;  // Wsum elements (bf16)

DEV float lrelu(float v) { return fmaxf(v, SLOPE * v); }

// A-operand element j in lane-half h maps to reduction index k = j + 16*h (matches B)
DEV int a_kmap(int j, int h) { return j + 16 * h; }
// B-operand element j in lane-half h maps to reduction index k = j + 16*h
// (contiguous k order used by this code: lanes 0-15 hold K=0..15, lanes 16-31 hold K=16..31)

// Deterministic per-wave stats reduction: each lane holds 8 channels (rows) of its
// pixel column; sum across the 16-lane half, lanes ln==0 publish to LDS; wave 0
// folds the 8 waves in fixed order and writes this workgroup's partial.
DEV void publish_stats(float ssum[2][8], float ssq[2][8],
                       float* sSum, float* sSq,   // [8][32] each in LDS
                       int wid, int ln, int h, int tid,
                       float* part /* [64] per wg */) {
#pragma unroll
  for (int mt = 0; mt < 2; ++mt)
#pragma unroll
    for (int r = 0; r < 8; ++r) {
      float s = ssum[mt][r], q = ssq[mt][r];
      for (int m = 1; m < 16; m <<= 1) {
        s += __shfl_xor(s, m, 32);
        q += __shfl_xor(q, m, 32);
      }
      if (ln == 0) {
        int c = mt * 16 + 8 * h + r;
        sSum[wid * 32 + c] = s;
        sSq [wid * 32 + c] = q;
      }
    }
  __syncthreads();
  if (tid < 32) {
    float s = 0.f, q = 0.f;
    for (int w = 0; w < 8; ++w) { s += sSum[w * 32 + tid]; q += sSq[w * 32 + tid]; }
    part[tid]      = s;
    part[32 + tid] = q;
  }
}

// ---------------------------------------------------------------------------
// K0: fold the H dimension of W1 into bf16 Wsum, stored in WMMA A-fragment
// order: element index = ((kb*2 + mt)*32 + lane)*16 + j, holding
// Wsum[m = mt*16 + (lane&15)][k = kb*32 + a_kmap(j, lane>>4)].
__global__ void k_prepW(const float* __restrict__ W1, __bf16* __restrict__ Ws) {
  int e = blockIdx.x * blockDim.x + threadIdx.x;
  if (e >= WS_ELEMS) return;
  int j  = e & 15;
  int L  = (e >> 4) & 31;
  int mt = (e >> 9) & 1;
  int kb = e >> 10;
  int h = L >> 4, ln = L & 15;
  int m  = mt * 16 + ln;
  int op = a_kmap(j, h);
  const float* base = W1 + (size_t)(m * 128 + kb * 4) * COUT + op;
  float s = base[0] + base[32] + base[64] + base[96];
  Ws[e] = (__bf16)s;
}

// ---------------------------------------------------------------------------
// K1: z0 = W0 @ y  (per 16-pixel tile, 2 M-tiles x 2 K-steps of WMMA)
__global__ void k_gemm1(const float* __restrict__ y, const float* __restrict__ W0,
                        float* __restrict__ z0, float* __restrict__ part) {
  __shared__ float sSum[8 * 32];
  __shared__ float sSq [8 * 32];
  int tid = threadIdx.x, wid = tid >> 5, lane = tid & 31, ln = lane & 15, h = lane >> 4;
  int gw = blockIdx.x * 8 + wid;
  int nw = gridDim.x * 8;

  BF16x16 A[2][2];
#pragma unroll
  for (int mt = 0; mt < 2; ++mt)
#pragma unroll
    for (int kb = 0; kb < 2; ++kb)
#pragma unroll
      for (int j = 0; j < 16; ++j) {
        int f = kb * 32 + a_kmap(j, h);
        A[mt][kb].e[j] = (__bf16)W0[(mt * 16 + ln) * CF + f];
      }

  float ssum[2][8] = {}, ssq[2][8] = {};

  for (int tile = gw; tile < P / 16; tile += nw) {
    int p = tile * 16 + ln;
    int b = p / NK, q = p - b * NK;
    const float* ycol = y + (size_t)b * CF * NK + q;
    __builtin_prefetch(ycol + (size_t)nw * 16, 0, 1);  // next tile (speculative)

    BF16x16 Bv[2];
#pragma unroll
    for (int kb = 0; kb < 2; ++kb)
#pragma unroll
      for (int j = 0; j < 16; ++j) {
        int f = kb * 32 + j + 16 * h;
        Bv[kb].e[j] = (__bf16)ycol[(size_t)f * NK];
      }

    v8f acc0 = {}, acc1 = {};
    acc0 = __builtin_amdgcn_wmma_f32_16x16x32_bf16(false, A[0][0].v, false, Bv[0].v, (short)0, acc0, false, false);
    acc0 = __builtin_amdgcn_wmma_f32_16x16x32_bf16(false, A[0][1].v, false, Bv[1].v, (short)0, acc0, false, false);
    acc1 = __builtin_amdgcn_wmma_f32_16x16x32_bf16(false, A[1][0].v, false, Bv[0].v, (short)0, acc1, false, false);
    acc1 = __builtin_amdgcn_wmma_f32_16x16x32_bf16(false, A[1][1].v, false, Bv[1].v, (short)0, acc1, false, false);

    float* zp = z0 + (size_t)p * COUT;
    f32x4 lo0 = {acc0[0], acc0[1], acc0[2], acc0[3]}, hi0 = {acc0[4], acc0[5], acc0[6], acc0[7]};
    f32x4 lo1 = {acc1[0], acc1[1], acc1[2], acc1[3]}, hi1 = {acc1[4], acc1[5], acc1[6], acc1[7]};
    *(f32x4*)(zp + 8 * h)          = lo0;  *(f32x4*)(zp + 8 * h + 4)      = hi0;
    *(f32x4*)(zp + 16 + 8 * h)     = lo1;  *(f32x4*)(zp + 16 + 8 * h + 4) = hi1;

#pragma unroll
    for (int r = 0; r < 8; ++r) {
      ssum[0][r] += acc0[r]; ssq[0][r] += acc0[r] * acc0[r];
      ssum[1][r] += acc1[r]; ssq[1][r] += acc1[r] * acc1[r];
    }
  }
  __syncthreads();
  publish_stats(ssum, ssq, sSum, sSq, wid, ln, h, tid, part + (size_t)blockIdx.x * 64);
}

// ---------------------------------------------------------------------------
// K2/K4/K6: fold per-WG partials -> BN scale/shift (fixed order, deterministic)
__global__ void k_bnstats(const float* __restrict__ part, int nwg,
                          const float* __restrict__ gamma, const float* __restrict__ beta,
                          float* __restrict__ scale, float* __restrict__ shift) {
  int c = threadIdx.x;  // 32 threads
  float s = 0.f, q = 0.f;
  for (int w = 0; w < nwg; ++w) { s += part[w * 64 + c]; q += part[w * 64 + 32 + c]; }
  float mean = s / (float)P;
  float var  = q / (float)P - mean * mean;
  float a = gamma[c] * rsqrtf(var + EPS);
  scale[c] = a;
  shift[c] = beta[c] - mean * a;
}

// ---------------------------------------------------------------------------
// K3: z1 = Wm @ lrelu(bn(z0))
__global__ void k_gemm2(const float* __restrict__ z0, const float* __restrict__ Wm,
                        const float* __restrict__ sc, const float* __restrict__ sh,
                        float* __restrict__ z1, float* __restrict__ part) {
  __shared__ float sSum[8 * 32];
  __shared__ float sSq [8 * 32];
  int tid = threadIdx.x, wid = tid >> 5, lane = tid & 31, ln = lane & 15, h = lane >> 4;
  int gw = blockIdx.x * 8 + wid;
  int nw = gridDim.x * 8;

  float a16[16], c16[16];
#pragma unroll
  for (int j = 0; j < 16; ++j) { a16[j] = sc[16 * h + j]; c16[j] = sh[16 * h + j]; }

  BF16x16 A[2];
#pragma unroll
  for (int mt = 0; mt < 2; ++mt)
#pragma unroll
    for (int j = 0; j < 16; ++j)
      A[mt].e[j] = (__bf16)Wm[(mt * 16 + ln) * COUT + a_kmap(j, h)];

  float ssum[2][8] = {}, ssq[2][8] = {};

  for (int tile = gw; tile < P / 16; tile += nw) {
    int p = tile * 16 + ln;
    const float* zp = z0 + (size_t)p * COUT + 16 * h;
    __builtin_prefetch(zp + (size_t)nw * 16 * COUT, 0, 1);  // next tile

    BF16x16 Bv;
#pragma unroll
    for (int j = 0; j < 16; ++j)
      Bv.e[j] = (__bf16)lrelu(a16[j] * zp[j] + c16[j]);

    v8f acc0 = {}, acc1 = {};
    acc0 = __builtin_amdgcn_wmma_f32_16x16x32_bf16(false, A[0].v, false, Bv.v, (short)0, acc0, false, false);
    acc1 = __builtin_amdgcn_wmma_f32_16x16x32_bf16(false, A[1].v, false, Bv.v, (short)0, acc1, false, false);

    float* op = z1 + (size_t)p * COUT;
    f32x4 lo0 = {acc0[0], acc0[1], acc0[2], acc0[3]}, hi0 = {acc0[4], acc0[5], acc0[6], acc0[7]};
    f32x4 lo1 = {acc1[0], acc1[1], acc1[2], acc1[3]}, hi1 = {acc1[4], acc1[5], acc1[6], acc1[7]};
    *(f32x4*)(op + 8 * h)          = lo0;  *(f32x4*)(op + 8 * h + 4)      = hi0;
    *(f32x4*)(op + 16 + 8 * h)     = lo1;  *(f32x4*)(op + 16 + 8 * h + 4) = hi1;

#pragma unroll
    for (int r = 0; r < 8; ++r) {
      ssum[0][r] += acc0[r]; ssq[0][r] += acc0[r] * acc0[r];
      ssum[1][r] += acc1[r]; ssq[1][r] += acc1[r] * acc1[r];
    }
  }
  __syncthreads();
  publish_stats(ssum, ssq, sSum, sSq, wid, ln, h, tid, part + (size_t)blockIdx.x * 64);
}

// ---------------------------------------------------------------------------
// K5: out_raw = Wsum(32x1024) @ q, q generated on the fly from x and t1.
// Wsum is in A-fragment order: lane L reads 32B at frag*1024 + L*32 -> two
// conflict-free ds_load_b128. Staged via async global->LDS when available.
__global__ void k_main(const float* __restrict__ z1, const float* __restrict__ x,
                       const __bf16* __restrict__ Ws,
                       const float* __restrict__ sc, const float* __restrict__ sh,
                       float* __restrict__ zo, float* __restrict__ part) {
  extern __shared__ char smem[];
  __bf16* sW   = (__bf16*)smem;                              // 64 KB, fragment order
  float*  sSum = (float*)(smem + WS_ELEMS * 2);              // 8*32
  float*  sSq  = sSum + 8 * 32;

  int tid = threadIdx.x, wid = tid >> 5, lane = tid & 31, ln = lane & 15, h = lane >> 4;

  // stage Wsum -> LDS (64 KB contiguous, b128 granularity)
#if USE_ASYNC_LDS
  {
    typedef __attribute__((address_space(1))) i32x4 as1_i32x4;
    typedef __attribute__((address_space(3))) i32x4 as3_i32x4;
    as1_i32x4* src = (as1_i32x4*)Ws;
    as3_i32x4* dst = (as3_i32x4*)sW;
    for (int i = tid; i < WS_ELEMS / 8; i += blockDim.x)
      __builtin_amdgcn_global_load_async_to_lds_b128(src + i, dst + i, 0, 0);
    __builtin_amdgcn_s_wait_asynccnt(0);
  }
#else
  {
    const u32x4* src = (const u32x4*)Ws;
    u32x4* dst = (u32x4*)sW;
    for (int i = tid; i < WS_ELEMS / 8; i += blockDim.x) dst[i] = src[i];
  }
#endif
  __syncthreads();

  float a16[16], c16[16];
#pragma unroll
  for (int j = 0; j < 16; ++j) { a16[j] = sc[16 * h + j]; c16[j] = sh[16 * h + j]; }

  int gw = blockIdx.x * 8 + wid;       // 0..1279 with grid 160
  float ssum[2][8] = {}, ssq[2][8] = {};

  for (int u = gw; u < P / 32; u += gridDim.x * 8) {
    int pbase = u * 32;

    float tf[2][16];
    const float* xb[2];
#pragma unroll
    for (int tt = 0; tt < 2; ++tt) {
      int p = pbase + tt * 16 + ln;
      int b = p / NK, q = p - b * NK;
      const float* zp = z1 + (size_t)p * COUT + 16 * h;
#pragma unroll
      for (int j = 0; j < 16; ++j) tf[tt][j] = lrelu(a16[j] * zp[j] + c16[j]);
      xb[tt] = x + (size_t)b * CIN * NK + q;
      __builtin_prefetch(xb[tt], 0, 1);
    }

    v8f acc[2][2] = {};

    for (int kb = 0; kb < 32; ++kb) {  // kb == input channel i
      const __bf16* f0 = sW + ((size_t)(kb * 2 + 0) * 32 + lane) * 16;
      const __bf16* f1 = sW + ((size_t)(kb * 2 + 1) * 32 + lane) * 16;
      BF16x16 A0, A1;
      A0.u[0] = *(const u32x4*)(f0);
      A0.u[1] = *(const u32x4*)(f0 + 8);
      A1.u[0] = *(const u32x4*)(f1);
      A1.u[1] = *(const u32x4*)(f1 + 8);
#pragma unroll
      for (int tt = 0; tt < 2; ++tt) {
        float xs = xb[tt][(size_t)kb * NK];
        BF16x16 Bv;
#pragma unroll
        for (int j = 0; j < 16; ++j) Bv.e[j] = (__bf16)(xs * tf[tt][j]);
        acc[0][tt] = __builtin_amdgcn_wmma_f32_16x16x32_bf16(false, A0.v, false, Bv.v, (short)0, acc[0][tt], false, false);
        acc[1][tt] = __builtin_amdgcn_wmma_f32_16x16x32_bf16(false, A1.v, false, Bv.v, (short)0, acc[1][tt], false, false);
      }
    }

#pragma unroll
    for (int tt = 0; tt < 2; ++tt) {
      int p = pbase + tt * 16 + ln;
      float* zp = zo + (size_t)p * COUT;
#pragma unroll
      for (int mt = 0; mt < 2; ++mt) {
        v8f a = acc[mt][tt];
        f32x4 lo = {a[0], a[1], a[2], a[3]}, hi = {a[4], a[5], a[6], a[7]};
        *(f32x4*)(zp + mt * 16 + 8 * h)     = lo;
        *(f32x4*)(zp + mt * 16 + 8 * h + 4) = hi;
#pragma unroll
        for (int r = 0; r < 8; ++r) { ssum[mt][r] += a[r]; ssq[mt][r] += a[r] * a[r]; }
      }
    }
  }
  __syncthreads();
  publish_stats(ssum, ssq, sSum, sSq, wid, ln, h, tid, part + (size_t)blockIdx.x * 64);
}

// ---------------------------------------------------------------------------
// K7: final = lrelu(bn(out_raw) + x), written channel-major (B, Cout, N, K)
__global__ void k_epilogue(const float* __restrict__ zo, const float* __restrict__ x,
                           const float* __restrict__ sc, const float* __restrict__ sh,
                           float* __restrict__ out) {
  int idx = blockIdx.x * blockDim.x + threadIdx.x;
  if (idx >= B_ * COUT * NK) return;
  int b   = idx / (COUT * NK);
  int rem = idx - b * COUT * NK;
  int o   = rem / NK;
  int q   = rem - o * NK;
  int p   = b * NK + q;
  float v = sc[o] * zo[(size_t)p * COUT + o] + sh[o] + x[idx];
  out[idx] = lrelu(v);
}

// ---------------------------------------------------------------------------
extern "C" void kernel_launch(void* const* d_in, const int* in_sizes, int n_in,
                              void* d_out, int out_size, void* d_ws, size_t ws_size,
                              hipStream_t stream) {
  const float* x     = (const float*)d_in[0];
  const float* y     = (const float*)d_in[1];
  const float* W0    = (const float*)d_in[2];
  const float* g0    = (const float*)d_in[3];
  const float* b0    = (const float*)d_in[4];
  const float* Wm    = (const float*)d_in[5];
  const float* gm    = (const float*)d_in[6];
  const float* bm    = (const float*)d_in[7];
  const float* W1    = (const float*)d_in[8];
  const float* g_out = (const float*)d_in[9];
  const float* b_out = (const float*)d_in[10];
  float* out = (float*)d_out;

  char* ws = (char*)d_ws;
  size_t off = 0;
  float*  z0   = (float*)(ws + off);  off += (size_t)P * COUT * sizeof(float);   // also reused as zo
  float*  z1   = (float*)(ws + off);  off += (size_t)P * COUT * sizeof(float);
  __bf16* Wsum = (__bf16*)(ws + off); off += (size_t)WS_ELEMS * sizeof(__bf16);
  float*  part = (float*)(ws + off);  off += (size_t)NWG_GEMM * 64 * sizeof(float);
  float*  a0c  = (float*)(ws + off);  off += 6 * 32 * sizeof(float);
  float* a0s = a0c, *c0s = a0c + 32, *a1s = a0c + 64, *c1s = a0c + 96, *a2s = a0c + 128, *c2s = a0c + 160;
  float* zo = z0;  // reuse: z0 is dead after K3

  (void)in_sizes; (void)n_in; (void)out_size; (void)ws_size;

  k_prepW<<<WS_ELEMS / 256, 256, 0, stream>>>(W1, Wsum);
  k_gemm1<<<NWG_GEMM, 256, 0, stream>>>(y, W0, z0, part);
  k_bnstats<<<1, 32, 0, stream>>>(part, NWG_GEMM, g0, b0, a0s, c0s);
  k_gemm2<<<NWG_GEMM, 256, 0, stream>>>(z0, Wm, a0s, c0s, z1, part);
  k_bnstats<<<1, 32, 0, stream>>>(part, NWG_GEMM, gm, bm, a1s, c1s);
  size_t smem = (size_t)WS_ELEMS * sizeof(__bf16) + 2 * 8 * 32 * sizeof(float);
  k_main<<<NWG_GEMM, 256, smem, stream>>>(z1, x, Wsum, a1s, c1s, zo, part);
  k_bnstats<<<1, 32, 0, stream>>>(part, NWG_GEMM, g_out, b_out, a2s, c2s);
  k_epilogue<<<(B_ * COUT * NK + 255) / 256, 256, 0, stream>>>(zo, x, a2s, c2s, out);
}
